// Upsample_MLP_multi_ResizeConvUp_16836271800449
// MI455X (gfx1250) — matrix operand order repaired and checked by reading, not compiled, not run on hardware
//
#include <hip/hip_runtime.h>
#include <hip/hip_bf16.h>

// ---------------------------------------------------------------------------
// Types & WMMA helpers (CDNA5 wave32, v_wmma_f32_16x16x32_bf16)
// ---------------------------------------------------------------------------
typedef __bf16 bf16t;
typedef __attribute__((ext_vector_type(16))) __bf16 bf16x16;
typedef __attribute__((ext_vector_type(8)))  __bf16 bf16x8;
typedef __attribute__((ext_vector_type(8)))  float  f32x8;

union FragU { bf16x16 v; struct { bf16x8 lo, hi; } p; };

// A fragment (16x32 bf16): lane group hi holds K = hi*8..hi*8+7 and +16..+23
__device__ __forceinline__ bf16x16 ldsFragA(const bf16t* tile, int rowBase,
                                            int rowStride, int k0) {
  const int lane = threadIdx.x & 31;
  const int r    = lane & 15;
  const int kb   = (lane >> 4) << 3;
  const bf16t* p = tile + (rowBase + r) * rowStride + k0 + kb;
  FragU u;
  u.p.lo = *(const bf16x8*)(p);
  u.p.hi = *(const bf16x8*)(p + 16);
  return u.v;
}

// B fragment (32x16 bf16): lane group hi holds K = hi*8..hi*8+7 and +16..+23
__device__ __forceinline__ bf16x16 ldsFragB(const bf16t* tile, int colBase,
                                            int rowStride, int k0) {
  const int lane = threadIdx.x & 31;
  const int n    = lane & 15;
  const int kb   = (lane >> 4) << 3;
  const bf16t* p = tile + (colBase + n) * rowStride + k0 + kb;
  FragU u;
  u.p.lo = *(const bf16x8*)(p);
  u.p.hi = *(const bf16x8*)(p + 16);
  return u.v;
}

__device__ __forceinline__ f32x8 wmma_bf16(bf16x16 a, bf16x16 b, f32x8 c) {
  return __builtin_amdgcn_wmma_f32_16x16x32_bf16(false, a, false, b,
                                                 (short)0, c, false, false);
}

// ---------------------------------------------------------------------------
// Async global -> LDS copy path (GLOBAL_LOAD_ASYNC_TO_LDS_B128, ASYNCcnt)
// Builtin signature (from probe diagnostics): arg0 = AS1 (global) int4*,
// arg1 = AS3 (LDS) int4*, then two int immediates (offset, cpol).
// ---------------------------------------------------------------------------
#if __has_builtin(__builtin_amdgcn_global_load_async_to_lds_b128)
#define HAS_ASYNC_LDS 1
#else
#define HAS_ASYNC_LDS 0
#endif

typedef int v4i_gcc __attribute__((vector_size(16)));
#define AS_GLOBAL __attribute__((address_space(1)))
#define AS_LDS    __attribute__((address_space(3)))

__device__ __forceinline__ void cp_async16(void* lds, const void* g) {
#if HAS_ASYNC_LDS
  __builtin_amdgcn_global_load_async_to_lds_b128(
      (AS_GLOBAL v4i_gcc*)g, (AS_LDS v4i_gcc*)lds, 0, 0);
#else
  *(uint4*)lds = *(const uint4*)g;
#endif
}

__device__ __forceinline__ void cp_async_wait() {
#if HAS_ASYNC_LDS
#if __has_builtin(__builtin_amdgcn_s_wait_asynccnt)
  __builtin_amdgcn_s_wait_asynccnt(0);
#else
  asm volatile("s_wait_asynccnt 0x0" ::: "memory");
#endif
#endif
}

// ---------------------------------------------------------------------------
// Workspace layout (bytes)
// ---------------------------------------------------------------------------
#define OFF_XU      0u
#define OFF_H1      268435456u   // xu: 8*256*256*256 bf16 = 268 MB
#define OFF_WBCONV  402653184u   // h1: 8*256*256*128 bf16 = 134 MB
#define OFF_WB1     403243008u   // wbconv: 9*128*256 bf16
#define OFF_WPACK   403316736u   // wb1: 128*288 bf16
#define OFF_SUM     403406848u   // wpack: 45056 bf16 (mlp3/4, [n][kpad])
#define OFF_SQ      403406880u
#define OFF_STATS   403406912u

// ---------------------------------------------------------------------------
// K0: bicubic 2x upsample, NCHW f32 -> BHWC bf16
// ---------------------------------------------------------------------------
__global__ __launch_bounds__(256) void k_upsample(const float* __restrict__ x,
                                                  bf16t* __restrict__ xu) {
  const float WE[4] = {-0.03515625f, 0.26171875f, 0.87890625f, -0.10546875f};
  int idx = blockIdx.x * 256 + threadIdx.x;       // 134,217,728 total
  int ox = idx & 255;
  int c  = (idx >> 8) & 255;
  int oy = (idx >> 16) & 255;
  int b  = idx >> 24;
  int kx = ox >> 1, px = ox & 1;
  int ky = oy >> 1, py = oy & 1;
  const float* xb = x + (b * 256 + c) * 128 * 128;
  float acc = 0.f;
#pragma unroll
  for (int my = 0; my < 4; ++my) {
    int ty = ky - 2 + py + my; ty = ty < 0 ? 0 : (ty > 127 ? 127 : ty);
    float wy = py ? WE[3 - my] : WE[my];
    float ra = 0.f;
#pragma unroll
    for (int mx = 0; mx < 4; ++mx) {
      int tx = kx - 2 + px + mx; tx = tx < 0 ? 0 : (tx > 127 ? 127 : tx);
      float wx = px ? WE[3 - mx] : WE[mx];
      ra += wx * xb[ty * 128 + tx];
    }
    acc += wy * ra;
  }
  xu[(((b * 256 + oy) * 256 + ox) << 8) + c] = (bf16t)acc;
}

// ---------------------------------------------------------------------------
// Weight repack: f32 [K][N] -> bf16 [N][Kpad] (zero pad), + conv repack
// ---------------------------------------------------------------------------
__global__ void k_pack_w(const float* __restrict__ src, bf16t* __restrict__ dst,
                         int K, int N, int Kpad) {
  int i = blockIdx.x * 256 + threadIdx.x;
  if (i >= N * Kpad) return;
  int n = i / Kpad, k = i - n * Kpad;
  dst[i] = (k < K) ? (bf16t)src[k * N + n] : (bf16t)0.f;
}

__global__ void k_pack_conv(const float* __restrict__ w, bf16t* __restrict__ dst) {
  int i = blockIdx.x * 256 + threadIdx.x;          // 9*128*256
  if (i >= 9 * 128 * 256) return;
  int tap = i / (128 * 256);
  int r   = i - tap * 128 * 256;
  int cout = r >> 8, cin = r & 255;
  dst[i] = (bf16t)w[(cout * 256 + cin) * 9 + tap]; // OIHW flat
}

__global__ void k_zero(float* p, int n) {
  int i = blockIdx.x * 256 + threadIdx.x;
  if (i < n) p[i] = 0.f;
}

// ---------------------------------------------------------------------------
// K1: fused conv3x3 (implicit GEMM over upsampled x) + concat(y,s) + MLP1+ReLU
//     tile = 8x16 spatial x 128 ch, 8 waves, wave = M16 row, N=128
//     xu halo tile double-buffered via async global->LDS copies
// ---------------------------------------------------------------------------
__device__ __forceinline__ void load_xu_tile_async(
    const bf16t* __restrict__ xu, bf16t* xuT, int b, int y0, int x0, int ck) {
  const int t = threadIdx.x;
  for (int p = t; p < 180; p += 256) {             // 10x18 halo, 32 cin each
    int ly = p / 18, lx = p - ly * 18;
    int gy = y0 - 1 + ly, gx = x0 - 1 + lx;
    bf16t* l = xuT + p * 32;
    if (gy >= 0 && gy < 256 && gx >= 0 && gx < 256) {
      const bf16t* g = xu + ((((b * 256 + gy) * 256 + gx) << 8) + ck * 32);
      cp_async16(l,      g);
      cp_async16(l + 8,  g + 8);
      cp_async16(l + 16, g + 16);
      cp_async16(l + 24, g + 24);
    } else {
      uint4 z = {0, 0, 0, 0};
      uint4* d = (uint4*)l;
      d[0] = z; d[1] = z; d[2] = z; d[3] = z;
    }
  }
}

__global__ __launch_bounds__(256) void k_conv_mlp1(
    const bf16t* __restrict__ xu, const bf16t* __restrict__ wbconv,
    const bf16t* __restrict__ wb1, const float* __restrict__ conv_b,
    const float* __restrict__ mlp1_b, const float* __restrict__ y,
    const float* __restrict__ s, bf16t* __restrict__ h1,
    float* __restrict__ sumBuf, float* __restrict__ sqBuf) {
  extern __shared__ char smem[];
  bf16t* xuT0 = (bf16t*)(smem);            // [180 pix][32 cin]   11520 B
  bf16t* xuT1 = (bf16t*)(smem + 11520);    // double buffer       11520 B
  bf16t* wbS  = (bf16t*)(smem + 23040);    // [128 cout][32 cin]   8192 B
  bf16t* A2   = (bf16t*)(smem + 31232);    // [128 pos][288 k]    73728 B
  bf16t* wb1L = (bf16t*)(smem + 104960);   // [128 n][288 k]      73728 B
  bf16t* h1T  = (bf16t*)(smem + 178688);   // [128 pos][128 c]    32768 B
  float* red  = (float*)(smem + 211456);   // 2 f32

  const int t = threadIdx.x, wave = t >> 5, lane = t & 31;
  const int cl = lane & 15, hi = lane >> 4;
  const int bidx = blockIdx.x;
  const int b  = bidx >> 9;                // 32*16 tiles per batch image
  const int tr = (bidx >> 4) & 31;
  const int tc = bidx & 15;
  const int y0 = tr * 8, x0 = tc * 16;

  // prefetch y rows used by the MLP1 A2-fill (overlaps conv phase)
  for (int i = t; i < 1024; i += 256) {
    int c = i >> 3, py = i & 7;
    __builtin_prefetch(y + ((b * 128 + c) * 256 + y0 + py) * 256 + x0, 0, 1);
  }

  // stage W1 (bf16 [128][288]) into LDS via async copies
  for (int i = t; i < 4608; i += 256)
    cp_async16(wb1L + i * 8, wb1 + i * 8);

  // kick off chunk-0 xu halo tile
  load_xu_tile_async(xu, xuT0, b, y0, x0, 0);

  f32x8 acc[8];
#pragma unroll
  for (int nt = 0; nt < 8; ++nt)
#pragma unroll
    for (int j = 0; j < 8; ++j) acc[nt][j] = 0.f;

  // ---- conv implicit GEMM: K = 8 chunks of 32 cin x 9 taps ----
  for (int ck = 0; ck < 8; ++ck) {
    cp_async_wait();                       // our async LDS writes landed
    __syncthreads();                       // tile visible to all waves
    bf16t* cur = (ck & 1) ? xuT1 : xuT0;
    if (ck < 7)                            // prefetch next chunk under compute
      load_xu_tile_async(xu, (ck & 1) ? xuT0 : xuT1, b, y0, x0, ck + 1);

    for (int tap = 0; tap < 9; ++tap) {
      const int dy = tap / 3 - 1, dx = tap - (tap / 3) * 3 - 1;
      __syncthreads();
      for (int i = t; i < 512; i += 256) { // 128 cout x 32 cin stage (sync path)
        int cout = i >> 2, g8 = i & 3;
        ((uint4*)wbS)[i] =
            *(const uint4*)(wbconv + (tap * 128 + cout) * 256 + ck * 32 + g8 * 8);
      }
      __syncthreads();
      bf16x16 a = ldsFragA(cur, (wave + dy + 1) * 18 + dx + 1, 32, 0);
#pragma unroll
      for (int nt = 0; nt < 8; ++nt)
        acc[nt] = wmma_bf16(a, ldsFragB(wbS, nt * 16, 32, 0), acc[nt]);
    }
  }

  // xc (+conv_b) -> A2[k=0..127] as bf16
#pragma unroll
  for (int nt = 0; nt < 8; ++nt) {
    float bb = conv_b[nt * 16 + cl];
#pragma unroll
    for (int rr = 0; rr < 8; ++rr)
      A2[(wave * 16 + rr + 8 * hi) * 288 + nt * 16 + cl] =
          (bf16t)(acc[nt][rr] + bb);
  }
  // y features -> A2[k=128..255]
  for (int i = t; i < 1024; i += 256) {        // (c, row) tasks, 16 x each
    int c = i >> 3, py = i & 7;
    const float* yr = y + ((b * 128 + c) * 256 + y0 + py) * 256 + x0;
#pragma unroll
    for (int px = 0; px < 16; ++px)
      A2[(py * 16 + px) * 288 + 128 + c] = (bf16t)yr[px];
  }
  const float sv = s[b];
  for (int p = t; p < 128; p += 256) {         // s at k=256, zero pad 257..287
    A2[p * 288 + 256] = (bf16t)sv;
    for (int k = 257; k < 288; ++k) A2[p * 288 + k] = (bf16t)0.f;
  }
  if (t < 2) red[t] = 0.f;
  __syncthreads();

  // ---- MLP1 GEMM: M16 x K288 x N128 per wave ----
#pragma unroll
  for (int nt = 0; nt < 8; ++nt)
#pragma unroll
    for (int j = 0; j < 8; ++j) acc[nt][j] = 0.f;
  for (int kc = 0; kc < 9; ++kc) {
    bf16x16 a = ldsFragA(A2, wave * 16, 288, kc * 32);
#pragma unroll
    for (int nt = 0; nt < 8; ++nt)
      acc[nt] = wmma_bf16(a, ldsFragB(wb1L, nt * 16, 288, kc * 32), acc[nt]);
  }
  float lsum = 0.f, lsq = 0.f;
#pragma unroll
  for (int nt = 0; nt < 8; ++nt) {
    float bb = mlp1_b[nt * 16 + cl];
#pragma unroll
    for (int rr = 0; rr < 8; ++rr) {
      float v = fmaxf(acc[nt][rr] + bb, 0.f);
      lsum += v; lsq += v * v;
      h1T[(wave * 16 + rr + 8 * hi) * 128 + nt * 16 + cl] = (bf16t)v;
    }
  }
  atomicAdd(&red[0], lsum);
  atomicAdd(&red[1], lsq);
  __syncthreads();
  if (t == 0) { atomicAdd(&sumBuf[b], red[0]); atomicAdd(&sqBuf[b], red[1]); }

  // contiguous h1 store (BHWC bf16), 128 B per thread
  {
    int pos = t >> 1, half = t & 1;
    int yy = y0 + (pos >> 4), xx = x0 + (pos & 15);
    const uint4* srcp = (const uint4*)(h1T + pos * 128 + half * 64);
    uint4* dstp =
        (uint4*)(h1 + (((b * 256 + yy) * 256 + xx) << 7) + half * 64);
#pragma unroll
    for (int i = 0; i < 8; ++i) dstp[i] = srcp[i];
  }
}

// ---------------------------------------------------------------------------
// K2: finalize per-batch LN statistics
// ---------------------------------------------------------------------------
__global__ void k_stats(const float* __restrict__ sumBuf,
                        const float* __restrict__ sqBuf,
                        float* __restrict__ stats) {
  int b = threadIdx.x;
  if (b < 8) {
    const float n = 256.f * 256.f * 128.f;
    float mu = sumBuf[b] / n;
    float var = sqBuf[b] / n - mu * mu;
    stats[2 * b]     = mu;
    stats[2 * b + 1] = rsqrtf(var + 1e-5f);
  }
}

// ---------------------------------------------------------------------------
// generic per-wave LDS GEMM step: D[pos][sD] = A[pos][sA] @ W^T (+bias,relu)
// ---------------------------------------------------------------------------
template <int NT, int KC>
__device__ __forceinline__ void gemm_tile(const bf16t* Abuf, int sA,
                                          const bf16t* Wbuf, int sW,
                                          const float* __restrict__ bias,
                                          bf16t* Dbuf, int sD, bool relu) {
  const int wave = threadIdx.x >> 5, lane = threadIdx.x & 31;
  const int cl = lane & 15, hi = lane >> 4;
  f32x8 acc[NT];
#pragma unroll
  for (int nt = 0; nt < NT; ++nt)
#pragma unroll
    for (int j = 0; j < 8; ++j) acc[nt][j] = 0.f;
#pragma unroll
  for (int kc = 0; kc < KC; ++kc) {
    bf16x16 a = ldsFragA(Abuf, wave * 16, sA, kc * 32);
#pragma unroll
    for (int nt = 0; nt < NT; ++nt)
      acc[nt] = wmma_bf16(a, ldsFragB(Wbuf, nt * 16, sW, kc * 32), acc[nt]);
  }
#pragma unroll
  for (int nt = 0; nt < NT; ++nt) {
    float bb = bias[nt * 16 + cl];
#pragma unroll
    for (int rr = 0; rr < 8; ++rr) {
      float v = acc[nt][rr] + bb;
      if (relu) v = fmaxf(v, 0.f);
      Dbuf[(wave * 16 + rr + 8 * hi) * sD + nt * 16 + cl] = (bf16t)v;
    }
  }
}

// ---------------------------------------------------------------------------
// K3: LN + MLP3 + MLP4 + residual, store NCHW f32
// ---------------------------------------------------------------------------
__global__ __launch_bounds__(256) void k_mlp34(
    const bf16t* __restrict__ h1, const bf16t* __restrict__ wpack,
    const float* __restrict__ s, const float* __restrict__ stats,
    const float* __restrict__ lnw, const float* __restrict__ lnb,
    const float* __restrict__ b31, const float* __restrict__ b32,
    const float* __restrict__ b33, const float* __restrict__ b41,
    const float* __restrict__ b42, const float* __restrict__ b43,
    float* __restrict__ out) {
  extern __shared__ char smem[];
  bf16t* h1T   = (bf16t*)(smem);            // [128][128]   32768 B
  bf16t* A3    = (bf16t*)(smem + 32768);    // [128][160]   40960 B
  bf16t* wm3_1 = (bf16t*)(smem + 73728);    // [64][160]
  bf16t* wm3_2 = (bf16t*)(smem + 94208);    // [64][64]
  bf16t* wm3_3 = (bf16t*)(smem + 102400);   // [128][64]
  bf16t* wm4_1 = (bf16t*)(smem + 118784);   // [64][160]
  bf16t* wm4_2 = (bf16t*)(smem + 139264);   // [64][64]
  bf16t* wm4_3 = (bf16t*)(smem + 147456);   // [128][64]
  bf16t* tB    = (bf16t*)(smem + 163840);   // [128][64]    16384 B
  bf16t* t2B   = (bf16t*)(smem + 180224);   // [128][64]    16384 B
  bf16t* uB    = (bf16t*)(smem + 196608);   // [128][160]   40960 B

  const int t = threadIdx.x, wave = t >> 5, lane = t & 31;
  const int cl = lane & 15, hi = lane >> 4;
  const int bidx = blockIdx.x;
  const int b  = bidx >> 9;
  const int tr = (bidx >> 4) & 31;
  const int tc = bidx & 15;
  const int y0 = tr * 8, x0 = tc * 16;

  // stage packed MLP weights (90112 B) + h1 tile via async copies
  for (int i = t; i < 5632; i += 256)
    cp_async16((bf16t*)wm3_1 + i * 8, wpack + i * 8);
  {
    int pos = t >> 1, half = t & 1;
    int yy = y0 + (pos >> 4), xx = x0 + (pos & 15);
    const bf16t* srcp = h1 + (((b * 256 + yy) * 256 + xx) << 7) + half * 64;
    bf16t* dstp = h1T + pos * 128 + half * 64;
#pragma unroll
    for (int i = 0; i < 8; ++i) cp_async16(dstp + i * 8, srcp + i * 8);
  }
  cp_async_wait();
  __syncthreads();

  const float mu = stats[2 * b], rstd = stats[2 * b + 1];
  const float sv = s[b];
  { // A3 = LN(h1)*ln_w + ln_b, cat s, zero-pad to 160
    int pos = t >> 1, half = t & 1;
    int yy = y0 + (pos >> 4), xx = x0 + (pos & 15);
    const float* w  = lnw + (((yy * 256) + xx) << 7) + half * 64;
    const float* bb = lnb + (((yy * 256) + xx) << 7) + half * 64;
#pragma unroll 8
    for (int i = 0; i < 64; ++i) {
      float h = (float)h1T[pos * 128 + half * 64 + i];
      A3[pos * 160 + half * 64 + i] = (bf16t)((h - mu) * rstd * w[i] + bb[i]);
    }
    if (half == 0) {
      A3[pos * 160 + 128] = (bf16t)sv;
      for (int k = 129; k < 160; ++k) A3[pos * 160 + k] = (bf16t)0.f;
    }
  }
  __syncthreads();

  // ---- MLP3: (129->64), (64->64), (64->128, relu) ----
  gemm_tile<4, 5>(A3, 160, wm3_1, 160, b31, tB, 64, false);
  __syncthreads();
  gemm_tile<4, 2>(tB, 64, wm3_2, 64, b32, t2B, 64, false);
  __syncthreads();
  gemm_tile<8, 2>(t2B, 64, wm3_3, 64, b33, uB, 160, true);
  if (t < 128) { // cat s + pad for MLP4 input
    uB[t * 160 + 128] = (bf16t)sv;
    for (int k = 129; k < 160; ++k) uB[t * 160 + k] = (bf16t)0.f;
  }
  __syncthreads();

  // ---- MLP4: (129->64), (64->64), (64->128 relu) + residual + store ----
  gemm_tile<4, 5>(uB, 160, wm4_1, 160, b41, tB, 64, false);
  __syncthreads();
  gemm_tile<4, 2>(tB, 64, wm4_2, 64, b42, t2B, 64, false);
  __syncthreads();
  {
    f32x8 acc[8];
#pragma unroll
    for (int nt = 0; nt < 8; ++nt)
#pragma unroll
      for (int j = 0; j < 8; ++j) acc[nt][j] = 0.f;
#pragma unroll
    for (int kc = 0; kc < 2; ++kc) {
      bf16x16 a = ldsFragA(t2B, wave * 16, 64, kc * 32);
#pragma unroll
      for (int nt = 0; nt < 8; ++nt)
        acc[nt] = wmma_bf16(a, ldsFragB(wm4_3, nt * 16, 64, kc * 32), acc[nt]);
    }
#pragma unroll
    for (int nt = 0; nt < 8; ++nt) {
      int cout = nt * 16 + cl;
      float bb = b43[cout];
      float vv[8];
#pragma unroll
      for (int rr = 0; rr < 8; ++rr) {
        float v = fmaxf(acc[nt][rr] + bb, 0.f);                  // relu(mlp4)
        v += (float)h1T[(wave * 16 + rr + 8 * hi) * 128 + cout]; // + identity
        vv[rr] = v;
      }
      // D rows rr -> x = x0 + 8*hi + rr : contiguous -> two float4 stores
      float* op = out + ((b * 128 + cout) * 256 + (y0 + wave)) * 256 + x0 + 8 * hi;
      *(float4*)(op)     = make_float4(vv[0], vv[1], vv[2], vv[3]);
      *(float4*)(op + 4) = make_float4(vv[4], vv[5], vv[6], vv[7]);
    }
  }
}

// ---------------------------------------------------------------------------
// Host launcher
// ---------------------------------------------------------------------------
extern "C" void kernel_launch(void* const* d_in, const int* in_sizes, int n_in,
                              void* d_out, int out_size, void* d_ws,
                              size_t ws_size, hipStream_t stream) {
  (void)in_sizes; (void)n_in; (void)out_size; (void)ws_size;
  const float* x      = (const float*)d_in[0];
  const float* y      = (const float*)d_in[1];
  const float* s      = (const float*)d_in[2];
  const float* conv_w = (const float*)d_in[4];
  const float* conv_b = (const float*)d_in[5];
  const float* mlp1_w = (const float*)d_in[6];
  const float* mlp1_b = (const float*)d_in[7];
  const float* ln_w   = (const float*)d_in[8];
  const float* ln_b   = (const float*)d_in[9];
  const float* m3_w1  = (const float*)d_in[10];
  const float* m3_b1  = (const float*)d_in[11];
  const float* m3_w2  = (const float*)d_in[12];
  const float* m3_b2  = (const float*)d_in[13];
  const float* m3_w3  = (const float*)d_in[14];
  const float* m3_b3  = (const float*)d_in[15];
  const float* m4_w1  = (const float*)d_in[16];
  const float* m4_b1  = (const float*)d_in[17];
  const float* m4_w2  = (const float*)d_in[18];
  const float* m4_b2  = (const float*)d_in[19];
  const float* m4_w3  = (const float*)d_in[20];
  const float* m4_b3  = (const float*)d_in[21];
  float* out = (float*)d_out;

  char* ws = (char*)d_ws;
  bf16t* xu     = (bf16t*)(ws + OFF_XU);
  bf16t* h1     = (bf16t*)(ws + OFF_H1);
  bf16t* wbconv = (bf16t*)(ws + OFF_WBCONV);
  bf16t* wb1    = (bf16t*)(ws + OFF_WB1);
  bf16t* wpack  = (bf16t*)(ws + OFF_WPACK);
  float* sumBuf = (float*)(ws + OFF_SUM);
  float* sqBuf  = (float*)(ws + OFF_SQ);
  float* stats  = (float*)(ws + OFF_STATS);

  // 0) zero reduction accumulators (every call -> deterministic)
  k_zero<<<1, 256, 0, stream>>>(sumBuf, 16);

  // 1) bicubic upsample -> xu (BHWC bf16)
  k_upsample<<<524288, 256, 0, stream>>>(x, xu);

  // 2) weight repacks -> bf16 [n][kpad]
  k_pack_conv<<<1152, 256, 0, stream>>>(conv_w, wbconv);
  k_pack_w<<<144, 256, 0, stream>>>(mlp1_w, wb1, 257, 128, 288);
  k_pack_w<<<40, 256, 0, stream>>>(m3_w1, wpack + 0,     129, 64, 160);
  k_pack_w<<<16, 256, 0, stream>>>(m3_w2, wpack + 10240,  64, 64,  64);
  k_pack_w<<<32, 256, 0, stream>>>(m3_w3, wpack + 14336,  64, 128, 64);
  k_pack_w<<<40, 256, 0, stream>>>(m4_w1, wpack + 22528, 129, 64, 160);
  k_pack_w<<<16, 256, 0, stream>>>(m4_w2, wpack + 32768,  64, 64,  64);
  k_pack_w<<<32, 256, 0, stream>>>(m4_w3, wpack + 36864,  64, 128, 64);

  // 3) fused conv + concat + MLP1 + LN stats (4096 tiles of 8x16 px)
  k_conv_mlp1<<<4096, 256, 211464, stream>>>(xu, wbconv, wb1, conv_b, mlp1_b,
                                             y, s, h1, sumBuf, sqBuf);

  // 4) finalize LN stats
  k_stats<<<1, 32, 0, stream>>>(sumBuf, sqBuf, stats);

  // 5) LN + MLP3 + MLP4 + residual + NCHW f32 out
  k_mlp34<<<4096, 256, 237568, stream>>>(h1, wpack, s, stats, ln_w, ln_b,
                                         m3_b1, m3_b2, m3_b3, m4_b1, m4_b2,
                                         m4_b3, out);
}
